// AttnReadout_without_intent_16655883174584
// MI455X (gfx1250) — matrix operand order repaired and checked by reading, not compiled, not run on hardware
//
#include <hip/hip_runtime.h>
#include <hip/hip_bf16.h>

// ---------------------------------------------------------------------------
// Problem constants (match reference)
// ---------------------------------------------------------------------------
#define N_NODES 1048576
#define N_SEG   16384
#define NPS     64          // nodes per segment
#define D_      128
#define EPS_    1e-5f

typedef __attribute__((ext_vector_type(16))) __bf16 v16bf;
typedef __attribute__((ext_vector_type(8)))  float  v8f;

// ---------------------------------------------------------------------------
// Workspace layout (bytes)
// ---------------------------------------------------------------------------
#define WS_SUM    0                     // 128 f32
#define WS_SSQ    512                   // 128 f32
#define WS_SCALE  1024                  // 128 f32
#define WS_SHIFT  1536                  // 128 f32
#define WS_WU16   2048                  // 128*128 bf16 (32 KB)
#define WS_WI16   34816                 // 128*128 bf16 (32 KB)
#define WS_FLAST  67584                 // 16384*128 f32 (8 MB)

// ---------------------------------------------------------------------------
// Kernel 1: per-column sum / sum-of-squares over all nodes
// grid 2048 x 256 threads; each block covers 512 rows; float4 column groups.
// ---------------------------------------------------------------------------
__global__ void k_stats(const float* __restrict__ feat,
                        float* __restrict__ sums, float* __restrict__ ssqs) {
  __shared__ float ls[8 * 128];
  __shared__ float lq[8 * 128];
  const int tid = threadIdx.x;
  const int c4  = (tid & 31) * 4;   // column group (4 cols)
  const int rg  = tid >> 5;         // row group 0..7
  const long row0 = (long)blockIdx.x * 512 + rg;
  float s0 = 0.f, s1 = 0.f, s2 = 0.f, s3 = 0.f;
  float q0 = 0.f, q1 = 0.f, q2 = 0.f, q3 = 0.f;
  for (int i = 0; i < 64; ++i) {
    const float4 v = *(const float4*)(feat + (row0 + (long)i * 8) * 128 + c4);
    s0 += v.x; q0 += v.x * v.x;
    s1 += v.y; q1 += v.y * v.y;
    s2 += v.z; q2 += v.z * v.z;
    s3 += v.w; q3 += v.w * v.w;
  }
  ls[rg * 128 + c4 + 0] = s0; lq[rg * 128 + c4 + 0] = q0;
  ls[rg * 128 + c4 + 1] = s1; lq[rg * 128 + c4 + 1] = q1;
  ls[rg * 128 + c4 + 2] = s2; lq[rg * 128 + c4 + 2] = q2;
  ls[rg * 128 + c4 + 3] = s3; lq[rg * 128 + c4 + 3] = q3;
  __syncthreads();
  if (tid < 128) {
    float a = 0.f;
    #pragma unroll
    for (int r = 0; r < 8; ++r) a += ls[r * 128 + tid];
    atomicAdd(&sums[tid], a);
  } else {
    const int d = tid - 128;
    float a = 0.f;
    #pragma unroll
    for (int r = 0; r < 8; ++r) a += lq[r * 128 + d];
    atomicAdd(&ssqs[d], a);
  }
}

// ---------------------------------------------------------------------------
// Kernel 2: fold BN into affine scale/shift per column
// ---------------------------------------------------------------------------
__global__ void k_finalize(const float* __restrict__ sums, const float* __restrict__ ssqs,
                           const float* __restrict__ gamma, const float* __restrict__ beta,
                           float* __restrict__ scale, float* __restrict__ shift) {
  const int d = threadIdx.x;
  const float inv_n = 1.f / (float)N_NODES;
  const float mean  = sums[d] * inv_n;
  float var = ssqs[d] * inv_n - mean * mean;
  var = fmaxf(var, 0.f);
  const float sc = gamma[d] * rsqrtf(var + EPS_);
  scale[d] = sc;
  shift[d] = beta[d] - mean * sc;
}

// ---------------------------------------------------------------------------
// Kernel 3: convert W_u / W_i to bf16 once (kept in L2 thereafter)
// ---------------------------------------------------------------------------
__global__ void k_convert(const float* __restrict__ wu, const float* __restrict__ wi,
                          __bf16* __restrict__ wu16, __bf16* __restrict__ wi16) {
  const int idx = blockIdx.x * 256 + threadIdx.x;
  if (idx < D_ * D_) {
    wu16[idx] = (__bf16)wu[idx];
    wi16[idx] = (__bf16)wi[idx];
  }
}

// ---------------------------------------------------------------------------
// Shared WMMA tile GEMM:
//   per wave: C[16x64] = A[16x128] * W^T  with A rows from fp32 LDS (cvt->bf16),
//   W (row-major [h][d]) as bf16 LDS.
// Layouts per CDNA5 ISA 7.12.2 (wave32):
//   A 16x32 bf16: lane&15 = M, lane>>4 selects K half (+16, contiguous K; ISA table uses +8/+16 split)
//   B 32x16 bf16: lane&15 = N, lane>>4 selects K half (+16, contiguous like A, not ISA order); pairs = consec K
//   C f32 16x16:  lane&15 = N, M = (lane>>4)*8 + vgpr
// ---------------------------------------------------------------------------
__device__ __forceinline__ void tile_gemm(const float* __restrict__ fb32,
                                          const __bf16* __restrict__ wb,
                                          int mt, int n_base, int lane, v8f acc[4]) {
  const int mn = lane & 15;
  const int hi = lane >> 4;
  const int m  = mt * 16 + mn;
  #pragma unroll
  for (int k = 0; k < 4; ++k) {
    const int ko = k * 32;
    const float* ap = fb32 + m * 128 + ko + hi * 16;
    v16bf a;
    #pragma unroll
    for (int j = 0; j < 8; ++j) a[j] = (__bf16)ap[j];        // K = ko + hi*16 + j
    #pragma unroll
    for (int j = 0; j < 8; ++j) a[j + 8] = (__bf16)ap[8 + j]; // K = ko+hi*16+8+j
    #pragma unroll
    for (int t = 0; t < 4; ++t) {
      const int h = n_base + t * 16 + mn;
      const v16bf b = *(const v16bf*)(wb + h * 128 + ko + hi * 16);
      acc[t] = __builtin_amdgcn_wmma_f32_16x16x32_bf16(
          false, a, false, b, (short)0, acc[t], false, false);
    }
  }
}

// ---------------------------------------------------------------------------
// Kernel 4: feat_last[seg][h] = f[last[seg]] @ W_i^T + b_i  (64 segs / block)
// dynamic LDS: fb32 32KB + wi bf16 32KB = 64KB
// ---------------------------------------------------------------------------
__global__ void k_featlast(const float* __restrict__ feat, const int* __restrict__ last,
                           const float* __restrict__ scale, const float* __restrict__ shift,
                           const __bf16* __restrict__ wi16, const float* __restrict__ b_i,
                           float* __restrict__ flast) {
  extern __shared__ char smem[];
  float*  fb32 = (float*)smem;
  __bf16* wb   = (__bf16*)(smem + 32768);
  const int tid = threadIdx.x;
  const int blk = blockIdx.x;

  // stage A: gather last-node rows, apply BN affine, keep fp32
  #pragma unroll
  for (int i = 0; i < 8; ++i) {
    const int idx = tid + 256 * i;          // 0..2047 float4s
    const int off = idx * 4;
    const int row = off >> 7;
    const int col = off & 127;
    const long g = (long)last[blk * 64 + row] * 128 + col;
    float4 v = *(const float4*)(feat + g);
    const float4 sc = *(const float4*)(scale + col);
    const float4 sh = *(const float4*)(shift + col);
    v.x = v.x * sc.x + sh.x; v.y = v.y * sc.y + sh.y;
    v.z = v.z * sc.z + sh.z; v.w = v.w * sc.w + sh.w;
    *(float4*)(fb32 + off) = v;
  }
  // stage W_i bf16 (32KB) via 16B chunks
  #pragma unroll
  for (int i = 0; i < 8; ++i) {
    const int idx = tid + 256 * i;          // 0..2047 uint4s
    ((uint4*)wb)[idx] = ((const uint4*)wi16)[idx];
  }
  __syncthreads();

  const int lane = tid & 31;
  const int wv   = tid >> 5;
  const int mt     = wv >> 1;
  const int n_base = (wv & 1) * 64;
  v8f acc[4];
  #pragma unroll
  for (int t = 0; t < 4; ++t)
    #pragma unroll
    for (int j = 0; j < 8; ++j) acc[t][j] = 0.f;
  tile_gemm(fb32, wb, mt, n_base, lane, acc);

  const int mn = lane & 15;
  const int hi = lane >> 4;
  #pragma unroll
  for (int t = 0; t < 4; ++t) {
    const int h  = n_base + t * 16 + mn;
    const float bi = b_i[h];
    #pragma unroll
    for (int r = 0; r < 8; ++r) {
      const int m = mt * 16 + hi * 8 + r;
      flast[((long)(blk * 64 + m)) * 128 + h] = acc[t][r] + bi;
    }
  }
}

// ---------------------------------------------------------------------------
// Kernel 5: main fused kernel, one block (8 waves) per segment.
//   - stage f (fp32, 32KB) and W_u (bf16, 32KB) -> exactly 64KB dynamic LDS
//   - WMMA GEMM (128 v_wmma per segment)
//   - gate e via sigmoid-dot(w_e) with shfl_xor + LDS float atomics
//     (e/alpha/pw buffers alias dead W_u region, barrier-separated)
//   - wave-0 softmax over 64 nodes
//   - fp32 weighted segment sums -> rst & position_rst
// ---------------------------------------------------------------------------
__global__ void k_main(const float* __restrict__ feat, const float* __restrict__ pw,
                       const float* __restrict__ scale, const float* __restrict__ shift,
                       const __bf16* __restrict__ wu16, const float* __restrict__ flast,
                       const float* __restrict__ w_e, float* __restrict__ out) {
  extern __shared__ char smem[];
  float*  fb32 = (float*)smem;                 // [64][128] fp32, live whole kernel
  __bf16* wb   = (__bf16*)(smem + 32768);      // [128][128] bf16, dead after GEMM
  float* e_buf  = (float*)(smem + 32768);      // aliases wb after GEMM barrier
  float* pw_buf = e_buf + 64;
  float* al_buf = pw_buf + 64;

  const int tid = threadIdx.x;
  const long seg = blockIdx.x;

  // stage f = feat*scale + shift (fp32)
  #pragma unroll
  for (int i = 0; i < 8; ++i) {
    const int idx = tid + 256 * i;
    const int off = idx * 4;
    const int row = off >> 7;
    const int col = off & 127;
    float4 v = *(const float4*)(feat + (seg * 64 + row) * 128 + col);
    const float4 sc = *(const float4*)(scale + col);
    const float4 sh = *(const float4*)(shift + col);
    v.x = v.x * sc.x + sh.x; v.y = v.y * sc.y + sh.y;
    v.z = v.z * sc.z + sh.z; v.w = v.w * sc.w + sh.w;
    *(float4*)(fb32 + off) = v;
  }
  // stage W_u bf16
  #pragma unroll
  for (int i = 0; i < 8; ++i) {
    const int idx = tid + 256 * i;
    ((uint4*)wb)[idx] = ((const uint4*)wu16)[idx];
  }
  __syncthreads();

  const int lane = tid & 31;
  const int wv   = tid >> 5;
  const int mt     = wv >> 1;          // 16-row tile: 0..3
  const int n_base = (wv & 1) * 64;    // 64 output cols per wave
  v8f acc[4];
  #pragma unroll
  for (int t = 0; t < 4; ++t)
    #pragma unroll
    for (int j = 0; j < 8; ++j) acc[t][j] = 0.f;
  tile_gemm(fb32, wb, mt, n_base, lane, acc);
  __syncthreads();                      // W_u LDS region now dead

  if (tid < 64) {
    e_buf[tid]  = 0.f;
    pw_buf[tid] = pw[seg * 64 + tid];
  }
  __syncthreads();

  // gate partials: e[m] += sum_h sigmoid(u + feat_last[h]) * w_e[h]
  const int mn = lane & 15;
  const int hi = lane >> 4;
  float part[8] = {0.f, 0.f, 0.f, 0.f, 0.f, 0.f, 0.f, 0.f};
  #pragma unroll
  for (int t = 0; t < 4; ++t) {
    const int h  = n_base + t * 16 + mn;
    const float fl  = flast[seg * 128 + h];
    const float wev = w_e[h];
    #pragma unroll
    for (int r = 0; r < 8; ++r) {
      const float u = acc[t][r] + fl;
      part[r] += wev / (1.f + __expf(-u));
    }
  }
  #pragma unroll
  for (int r = 0; r < 8; ++r) {
    part[r] += __shfl_xor(part[r], 1, 32);
    part[r] += __shfl_xor(part[r], 2, 32);
    part[r] += __shfl_xor(part[r], 4, 32);
    part[r] += __shfl_xor(part[r], 8, 32);
  }
  if (mn == 0) {
    #pragma unroll
    for (int r = 0; r < 8; ++r)
      atomicAdd(&e_buf[mt * 16 + hi * 8 + r], part[r]);
  }
  __syncthreads();

  // softmax over the 64 nodes (wave 0)
  if (tid < 32) {
    const float e0 = e_buf[tid];
    const float e1 = e_buf[tid + 32];
    float mx = fmaxf(e0, e1);
    #pragma unroll
    for (int msk = 16; msk >= 1; msk >>= 1)
      mx = fmaxf(mx, __shfl_xor(mx, msk, 32));
    const float x0 = __expf(e0 - mx);
    const float x1 = __expf(e1 - mx);
    float sm = x0 + x1;
    #pragma unroll
    for (int msk = 16; msk >= 1; msk >>= 1)
      sm += __shfl_xor(sm, msk, 32);
    const float inv = 1.f / sm;
    al_buf[tid]      = x0 * inv;
    al_buf[tid + 32] = x1 * inv;
  }
  __syncthreads();

  // weighted segment sums (fp32): tid<128 -> rst, tid>=128 -> position_rst
  const int d     = tid & 127;
  const int which = tid >> 7;
  const float* wsel = which ? pw_buf : al_buf;
  float a0 = 0.f;
  #pragma unroll 8
  for (int n = 0; n < 64; ++n) a0 += wsel[n] * fb32[n * 128 + d];
  out[(long)which * ((long)N_SEG * 128) + seg * 128 + d] = a0;
}

// ---------------------------------------------------------------------------
// Host launcher
// ---------------------------------------------------------------------------
extern "C" void kernel_launch(void* const* d_in, const int* in_sizes, int n_in,
                              void* d_out, int out_size, void* d_ws, size_t ws_size,
                              hipStream_t stream) {
  const float* feat  = (const float*)d_in[0];
  const float* pw    = (const float*)d_in[1];
  const int*   last  = (const int*)d_in[2];
  // d_in[3] segment_ids: implied by contiguous equal segments
  const float* gamma = (const float*)d_in[4];
  const float* beta  = (const float*)d_in[5];
  const float* W_u   = (const float*)d_in[6];
  const float* W_i   = (const float*)d_in[7];
  const float* b_i   = (const float*)d_in[8];
  const float* w_e   = (const float*)d_in[9];

  char* ws = (char*)d_ws;
  float*  sums  = (float*)(ws + WS_SUM);
  float*  ssqs  = (float*)(ws + WS_SSQ);
  float*  scale = (float*)(ws + WS_SCALE);
  float*  shift = (float*)(ws + WS_SHIFT);
  __bf16* wu16  = (__bf16*)(ws + WS_WU16);
  __bf16* wi16  = (__bf16*)(ws + WS_WI16);
  float*  flast = (float*)(ws + WS_FLAST);
  float*  out   = (float*)d_out;

  hipMemsetAsync(d_ws, 0, 1024, stream);  // zero sum/sumsq accumulators

  hipLaunchKernelGGL(k_stats,    dim3(2048),     dim3(256), 0,     stream, feat, sums, ssqs);
  hipLaunchKernelGGL(k_finalize, dim3(1),        dim3(128), 0,     stream, sums, ssqs, gamma, beta, scale, shift);
  hipLaunchKernelGGL(k_convert,  dim3(64),       dim3(256), 0,     stream, W_u, W_i, wu16, wi16);
  hipLaunchKernelGGL(k_featlast, dim3(N_SEG/64), dim3(256), 65536, stream, feat, last, scale, shift, wi16, b_i, flast);
  hipLaunchKernelGGL(k_main,     dim3(N_SEG),    dim3(256), 65536, stream, feat, pw, scale, shift, wu16, flast, w_e, out);
}
